// RPN_34136400068867
// MI455X (gfx1250) — matrix operand order repaired and checked by reading, not compiled, not run on hardware
//
#include <hip/hip_runtime.h>
#include <hip/hip_bf16.h>

#define HW 64
#define HWP 66
#define C_IN 512
#define C_MID 512
#define KTOT 4608          // 9 * 512, K order = (ky*3+kx)*512 + ic
#define MPB 4096           // 64*64 spatial per batch
#define NPROP 36864        // 4096 * 9 anchors
#define PRE_N 2000
#define POST_N 300
#define NEG_INF -1e30f
#define NBIN 8192

typedef __attribute__((ext_vector_type(16))) __bf16 v16bf;
typedef __attribute__((ext_vector_type(8)))  float  v8f;

struct alignas(16) Quad { unsigned v[4]; };

union Frag16 {
    unsigned short us[16];
    Quad q[2];
    v16bf v;
};

__device__ __forceinline__ unsigned short f2bf(float f) {
    unsigned u = __float_as_uint(f);
    unsigned r = u + 0x7fffu + ((u >> 16) & 1u);   // round-to-nearest-even
    return (unsigned short)(r >> 16);
}
__device__ __forceinline__ float bf2f(unsigned short h) {
    return __uint_as_float(((unsigned)h) << 16);
}

// canonical generate_anchors(16, (0.5,1,2), (8,16,32)) output
__constant__ float c_anch[9][4] = {
    {-84.f,-40.f,99.f,55.f},   {-176.f,-88.f,191.f,103.f}, {-360.f,-184.f,375.f,199.f},
    {-56.f,-56.f,71.f,71.f},   {-120.f,-120.f,135.f,135.f},{-248.f,-248.f,263.f,263.f},
    {-36.f,-80.f,51.f,95.f},   {-80.f,-168.f,95.f,183.f},  {-168.f,-344.f,183.f,359.f}};

// ---------------- zero-fill channel-last padded input ----------------
__global__ void k_zero(Quad* __restrict__ p, long nq) {
    long i = (long)blockIdx.x * 256 + threadIdx.x;
    if (i < nq) {
        Quad z; z.v[0] = z.v[1] = z.v[2] = z.v[3] = 0u;
        p[i] = z;
    }
}

// ---------------- transpose NCHW fp32 -> padded channel-last bf16 [B][66][66][512] ----------------
__global__ __launch_bounds__(256) void k_tr(const float* __restrict__ in,
                                            unsigned short* __restrict__ inT) {
    __shared__ unsigned short tile[64 * 72];   // [c][x], padded rows
    int bx = blockIdx.x;
    int b = bx >> 6, y = bx & 63;
    int t = threadIdx.x;
    int q  = t >> 2;            // read: channel ; write: x
    int s4 = (t & 3) << 4;      // 0,16,32,48
    for (int c0 = 0; c0 < C_IN; c0 += 64) {
        const float* src = in + (((long)(b * C_IN + c0 + q) * HW + y) * HW + s4);
        union { unsigned short us[16]; Quad qq[2]; } pk;
        #pragma unroll
        for (int j = 0; j < 16; ++j) pk.us[j] = f2bf(src[j]);
        *(Quad*)&tile[q * 72 + s4]     = pk.qq[0];
        *(Quad*)&tile[q * 72 + s4 + 8] = pk.qq[1];
        __syncthreads();
        union { unsigned short us[16]; Quad qq[2]; } po;
        #pragma unroll
        for (int j = 0; j < 16; ++j) po.us[j] = tile[(s4 + j) * 72 + q];
        unsigned short* dst = inT + (((long)(b * HWP + y + 1) * HWP) + (q + 1)) * C_IN + c0 + s4;
        *(Quad*)dst       = po.qq[0];
        *(Quad*)(dst + 8) = po.qq[1];
        __syncthreads();
    }
}

// ---------------- pre-pass: weight repack [oc][(tap,ic)] bf16 ----------------
__global__ void k_w1(const float* __restrict__ w, unsigned short* __restrict__ o, long total) {
    long e = (long)blockIdx.x * 256 + threadIdx.x;
    if (e >= total) return;
    int k  = (int)(e % KTOT);
    int oc = (int)(e / KTOT);
    int pos = k / C_IN, ic = k % C_IN;
    int ky = pos / 3, kx = pos % 3;
    o[e] = f2bf(w[(((long)oc * C_IN + ic) * 3 + ky) * 3 + kx]);
}

// ---------------- 3x3 conv + ReLU: implicit GEMM, all fragments straight from global ----------------
__global__ __launch_bounds__(256) void k_conv1(
    const unsigned short* __restrict__ inT,   // bf16 [B][66][66][512], zero borders
    const unsigned short* __restrict__ w1,    // bf16 [512][4608]
    const float* __restrict__ conv_b,
    unsigned short* __restrict__ xout)        // bf16 [B][512][4096]
{
    int bx = blockIdx.x;
    int mt = bx >> 3, nt = bx & 7;
    int b   = mt >> 5;
    int mb0 = (mt & 31) << 7;                 // spatial base within batch
    int t = threadIdx.x, lane = t & 31, wid = t >> 5;
    int wm = wid & 3, wn = wid >> 2;
    int sel = lane >> 4;

    v8f zero = {0.f,0.f,0.f,0.f,0.f,0.f,0.f,0.f};
    v8f acc[2][2];
    acc[0][0] = zero; acc[0][1] = zero; acc[1][0] = zero; acc[1][1] = zero;

    // per-lane fragment base offsets (halfs)
    long abase[2];
    #pragma unroll
    for (int fm = 0; fm < 2; ++fm) {
        int m  = mb0 + wm * 32 + fm * 16 + (lane & 15);
        int gy = m >> 6, gx = m & 63;
        abase[fm] = (((long)(b * HWP + gy) * HWP) + gx) * C_IN + sel * 8;
    }
    long bbase[2];
    #pragma unroll
    for (int fn = 0; fn < 2; ++fn) {
        int n = nt * 64 + wn * 32 + fn * 16 + (lane & 15);
        bbase[fn] = (long)n * KTOT + sel * 8;
    }

    for (int tap = 0; tap < 9; ++tap) {
        int ky = tap / 3, kx = tap % 3;
        const unsigned short* ap0 = inT + abase[0] + (long)(ky * HWP + kx) * C_IN;
        const unsigned short* ap1 = inT + abase[1] + (long)(ky * HWP + kx) * C_IN;
        const unsigned short* bp0 = w1 + bbase[0] + tap * C_IN;
        const unsigned short* bp1 = w1 + bbase[1] + tap * C_IN;

        if (tap + 1 < 9) {   // prefetch next tap's A rows (global_prefetch_b8)
            int ny = (tap + 1) / 3, nx = (tap + 1) % 3;
            __builtin_prefetch(inT + abase[0] + (long)(ny * HWP + nx) * C_IN, 0, 1);
        }

        #pragma unroll 2
        for (int icb = 0; icb < C_IN; icb += 32) {
            Frag16 fa0, fa1, fb0, fb1;
            // A: lane = row m, k-local = sel*8..+7 and sel*8+16..+23 (contiguous ic)
            fa0.q[0] = *(const Quad*)(ap0 + icb);
            fa0.q[1] = *(const Quad*)(ap0 + icb + 16);
            fa1.q[0] = *(const Quad*)(ap1 + icb);
            fa1.q[1] = *(const Quad*)(ap1 + icb + 16);
            // B: lane = col n, k-local = sel*8..+7 and sel*8+16..+23 (contiguous ic)
            fb0.q[0] = *(const Quad*)(bp0 + icb);
            fb0.q[1] = *(const Quad*)(bp0 + icb + 16);
            fb1.q[0] = *(const Quad*)(bp1 + icb);
            fb1.q[1] = *(const Quad*)(bp1 + icb + 16);

            acc[0][0] = __builtin_amdgcn_wmma_f32_16x16x32_bf16(
                false, fa0.v, false, fb0.v, (short)0, acc[0][0], false, false);
            acc[0][1] = __builtin_amdgcn_wmma_f32_16x16x32_bf16(
                false, fa0.v, false, fb1.v, (short)0, acc[0][1], false, false);
            acc[1][0] = __builtin_amdgcn_wmma_f32_16x16x32_bf16(
                false, fa1.v, false, fb0.v, (short)0, acc[1][0], false, false);
            acc[1][1] = __builtin_amdgcn_wmma_f32_16x16x32_bf16(
                false, fa1.v, false, fb1.v, (short)0, acc[1][1], false, false);
        }
    }

    // epilogue: +bias, ReLU, bf16 pack, one b128 store per fragment per lane
    #pragma unroll
    for (int fm = 0; fm < 2; ++fm)
        #pragma unroll
        for (int fn = 0; fn < 2; ++fn) {
            int N = nt * 64 + wn * 32 + fn * 16 + (lane & 15);
            float bias = conv_b[N];
            int mstore = mb0 + wm * 32 + fm * 16 + sel * 8;
            union { unsigned short us[8]; Quad q; } pk;
            #pragma unroll
            for (int r = 0; r < 8; ++r) {
                float v = acc[fm][fn][r] + bias;
                pk.us[r] = f2bf(v > 0.f ? v : 0.f);
            }
            *(Quad*)&xout[((long)(b * C_MID + N)) * MPB + mstore] = pk.q;
        }
}

// ---------------- 1x1 convs + sigmoid-softmax + anchor decode ----------------
__global__ __launch_bounds__(256) void k_conv2(
    const unsigned short* __restrict__ xbf,
    const float* __restrict__ cls_w, const float* __restrict__ cls_b,
    const float* __restrict__ bbox_w, const float* __restrict__ bbox_b,
    const float* __restrict__ im_info,
    float* __restrict__ scores, float* __restrict__ boxes)
{
    __shared__ unsigned short ldsX[64 * 256];  // 64 channels x 256 m
    int bx = blockIdx.x;
    int b  = bx >> 4;
    int m0 = (bx & 15) << 8;
    int t = threadIdx.x, lane = t & 31, wid = t >> 5;

    float acc[54];
    #pragma unroll
    for (int i = 0; i < 54; ++i) acc[i] = 0.f;

    for (int c0 = 0; c0 < C_MID; c0 += 64) {
        #pragma unroll
        for (int rr = 0; rr < 8; ++rr) {
            int cc = wid * 8 + rr;
            *(Quad*)&ldsX[cc * 256 + lane * 8] =
                *(const Quad*)&xbf[((long)(b * C_MID + c0 + cc)) * MPB + m0 + lane * 8];
        }
        __syncthreads();
        for (int cc = 0; cc < 64; ++cc) {
            float xv = bf2f(ldsX[cc * 256 + t]);
            int c = c0 + cc;
            #pragma unroll
            for (int oc = 0; oc < 18; ++oc) acc[oc]      += xv * cls_w[oc * C_MID + c];
            #pragma unroll
            for (int oc = 0; oc < 36; ++oc) acc[18 + oc] += xv * bbox_w[oc * C_MID + c];
        }
        __syncthreads();
    }

    int mm = m0 + t;
    float Hh = im_info[b * 3 + 0], Ww = im_info[b * 3 + 1], sc = im_info[b * 3 + 2];
    float mins = 16.f * sc;
    int gy = mm >> 6, gx = mm & 63;
    float shx = gx * 16.f, shy = gy * 16.f;

    #pragma unroll
    for (int a = 0; a < 9; ++a) {
        float s0 = acc[a] + cls_b[a];
        float s1 = acc[9 + a] + cls_b[9 + a];
        float sfg = 1.f / (1.f + expf(s0 - s1));   // softmax over {bg,fg}

        float dx = acc[18 + 4 * a + 0] + bbox_b[4 * a + 0];
        float dy = acc[18 + 4 * a + 1] + bbox_b[4 * a + 1];
        float dw = acc[18 + 4 * a + 2] + bbox_b[4 * a + 2];
        float dh = acc[18 + 4 * a + 3] + bbox_b[4 * a + 3];

        float ax1 = c_anch[a][0] + shx, ay1 = c_anch[a][1] + shy;
        float ax2 = c_anch[a][2] + shx, ay2 = c_anch[a][3] + shy;
        float wa = ax2 - ax1 + 1.f, ha = ay2 - ay1 + 1.f;
        float cxa = ax1 + 0.5f * wa, cya = ay1 + 0.5f * ha;
        float cx = dx * wa + cxa, cy = dy * ha + cya;
        float pw = expf(dw) * wa, ph = expf(dh) * ha;
        float x1 = fminf(fmaxf(cx - 0.5f * pw, 0.f), Ww - 1.f);
        float y1 = fminf(fmaxf(cy - 0.5f * ph, 0.f), Hh - 1.f);
        float x2 = fminf(fmaxf(cx + 0.5f * pw, 0.f), Ww - 1.f);
        float y2 = fminf(fmaxf(cy + 0.5f * ph, 0.f), Hh - 1.f);
        bool keep = (x2 - x1 + 1.f >= mins) && (y2 - y1 + 1.f >= mins);

        long p = (long)b * NPROP + (long)mm * 9 + a;
        scores[p] = keep ? sfg : NEG_INF;
        boxes[p * 4 + 0] = x1; boxes[p * 4 + 1] = y1;
        boxes[p * 4 + 2] = x2; boxes[p * 4 + 3] = y2;
    }
}

// ---------------- per-batch histogram -> threshold bin with >=2000 above ----------------
__global__ __launch_bounds__(1024) void k_hist(const float* __restrict__ scores,
                                               int* __restrict__ T, int* __restrict__ candCount) {
    __shared__ int hist[NBIN];
    __shared__ int segsum[1024];
    __shared__ int segsuf[1024];
    __shared__ int Tsh;
    int b = blockIdx.x, t = threadIdx.x;
    for (int i = t; i < NBIN; i += 1024) hist[i] = 0;
    if (t == 0) Tsh = 0;
    __syncthreads();
    const float* s = scores + (long)b * NPROP;
    for (int r = 0; r < 36; ++r) {
        float v = s[t + r * 1024] * (float)NBIN;
        int bin = v <= 0.f ? 0 : (v >= (float)(NBIN - 1) ? NBIN - 1 : (int)v);
        atomicAdd(&hist[bin], 1);
    }
    __syncthreads();
    int ss = 0;
    for (int i = 0; i < 8; ++i) ss += hist[t * 8 + i];
    segsum[t] = ss;
    __syncthreads();
    if (t == 0) {
        int c = 0;
        for (int i = 1023; i >= 0; --i) { segsuf[i] = c; c += segsum[i]; }
    }
    __syncthreads();
    int cum = segsuf[t];
    int localT = -1;
    for (int bin = t * 8 + 7; bin >= t * 8; --bin) {
        cum += hist[bin];
        if (cum >= PRE_N) { localT = bin; break; }
    }
    if (localT >= 0) atomicMax(&Tsh, localT);
    __syncthreads();
    if (t == 0) { T[b] = Tsh; candCount[b] = 0; }
}

__global__ void k_compact(const float* __restrict__ scores, const int* __restrict__ T,
                          int* __restrict__ candCount, int* __restrict__ cand) {
    int b   = blockIdx.x / 144;
    int blk = blockIdx.x % 144;
    int p = blk * 256 + threadIdx.x;
    float v = scores[(long)b * NPROP + p] * (float)NBIN;
    int bin = v <= 0.f ? 0 : (v >= (float)(NBIN - 1) ? NBIN - 1 : (int)v);
    if (bin >= T[b]) {
        int pos = atomicAdd(&candCount[b], 1);
        cand[(long)b * NPROP + pos] = p;
    }
}

// exact rank among candidates (score desc, index asc) -> sorted top-2000
__global__ __launch_bounds__(256) void k_rank(
    const float* __restrict__ scores, const float* __restrict__ boxes,
    const int* __restrict__ candCount, const int* __restrict__ cand,
    float* __restrict__ topScore, float* __restrict__ topBox) {
    __shared__ float sS[256];
    __shared__ int   sI[256];
    int b   = blockIdx.x / 144;
    int blk = blockIdx.x % 144;
    int t = threadIdx.x;
    int C = candCount[b];
    int gi = blk * 256 + t;
    int myIdx = 0; float myS = 0.f;
    if (gi < C) { myIdx = cand[(long)b * NPROP + gi]; myS = scores[(long)b * NPROP + myIdx]; }
    int rank = 0;
    for (int j0 = 0; j0 < C; j0 += 256) {
        int jj = j0 + t;
        if (jj < C) {
            int id = cand[(long)b * NPROP + jj];
            sI[t] = id; sS[t] = scores[(long)b * NPROP + id];
        }
        __syncthreads();
        int lim = C - j0; if (lim > 256) lim = 256;
        if (gi < C) {
            for (int k = 0; k < lim; ++k) {
                float sj = sS[k]; int ij = sI[k];
                rank += (sj > myS) || (sj == myS && ij < myIdx);
            }
        }
        __syncthreads();
    }
    if (gi < C && rank < PRE_N) {
        topScore[(long)b * PRE_N + rank] = myS;
        long src = ((long)b * NPROP + myIdx) * 4;
        long dst = ((long)b * PRE_N + rank) * 4;
        topBox[dst + 0] = boxes[src + 0]; topBox[dst + 1] = boxes[src + 1];
        topBox[dst + 2] = boxes[src + 2]; topBox[dst + 3] = boxes[src + 3];
    }
}

// ---------------- 2000x2000 IoU suppression bit rows ----------------
__global__ __launch_bounds__(256) void k_iou(const float* __restrict__ topBox,
                                             unsigned* __restrict__ rowmask) {
    __shared__ float sx1[PRE_N], sy1[PRE_N], sx2[PRE_N], sy2[PRE_N];
    int b = blockIdx.x >> 3, blk = blockIdx.x & 7;
    int t = threadIdx.x;
    const float* tb = topBox + (long)b * PRE_N * 4;
    for (int j = t; j < PRE_N; j += 256) {
        sx1[j] = tb[j * 4 + 0]; sy1[j] = tb[j * 4 + 1];
        sx2[j] = tb[j * 4 + 2]; sy2[j] = tb[j * 4 + 3];
    }
    __syncthreads();
    int i = blk * 256 + t;
    if (i >= PRE_N) return;
    float x1 = sx1[i], y1 = sy1[i], x2 = sx2[i], y2 = sy2[i];
    float ai = (x2 - x1 + 1.f) * (y2 - y1 + 1.f);
    unsigned* row = rowmask + ((long)b * PRE_N + i) * 63;
    for (int w = 0; w < 63; ++w) {
        unsigned bits = 0u;
        #pragma unroll 4
        for (int jb = 0; jb < 32; ++jb) {
            int j = w * 32 + jb;
            if (j < PRE_N && j > i) {
                float xx1 = fmaxf(x1, sx1[j]), yy1 = fmaxf(y1, sy1[j]);
                float xx2 = fminf(x2, sx2[j]), yy2 = fminf(y2, sy2[j]);
                float iw = fmaxf(xx2 - xx1 + 1.f, 0.f);
                float ih = fmaxf(yy2 - yy1 + 1.f, 0.f);
                float inter = iw * ih;
                float aj = (sx2[j] - sx1[j] + 1.f) * (sy2[j] - sy1[j] + 1.f);
                if (inter / (ai + aj - inter) > 0.7f) bits |= 1u << jb;
            }
        }
        row[w] = bits;
    }
}

// ---------------- serial NMS scan + emit first 300 rois ----------------
__global__ void k_scan(const unsigned* __restrict__ rowmask,
                       const float* __restrict__ topScore,
                       const float* __restrict__ topBox,
                       float* __restrict__ out) {
    __shared__ unsigned rem[63];
    __shared__ int flag;
    int b = blockIdx.x, t = threadIdx.x;
    if (t < 63) rem[t] = 0u;
    __syncthreads();
    const unsigned* rm = rowmask + (long)b * PRE_N * 63;
    for (int i = 0; i < PRE_N; ++i) {
        if (t == 0) flag = (int)((rem[i >> 5] >> (i & 31)) & 1u);
        __syncthreads();
        if (!flag && t < 63) rem[t] |= rm[(long)i * 63 + t];
        __syncthreads();
    }
    if (t == 0) {
        const float* ts = topScore + (long)b * PRE_N;
        const float* tb = topBox + (long)b * PRE_N * 4;
        float* o = out + (long)b * POST_N * 5;
        int slot = 0;
        for (int i = 0; i < PRE_N && slot < POST_N; ++i) {
            bool sup = (rem[i >> 5] >> (i & 31)) & 1u;
            if (!sup && ts[i] > -1e29f) {
                o[slot * 5 + 0] = (float)b;
                o[slot * 5 + 1] = tb[i * 4 + 0]; o[slot * 5 + 2] = tb[i * 4 + 1];
                o[slot * 5 + 3] = tb[i * 4 + 2]; o[slot * 5 + 4] = tb[i * 4 + 3];
                ++slot;
            }
        }
        for (int i = 0; i < PRE_N && slot < POST_N; ++i) {
            bool sup = (rem[i >> 5] >> (i & 31)) & 1u;
            if (sup || ts[i] <= -1e29f) {
                o[slot * 5 + 0] = (float)b;
                o[slot * 5 + 1] = tb[i * 4 + 0]; o[slot * 5 + 2] = tb[i * 4 + 1];
                o[slot * 5 + 3] = tb[i * 4 + 2]; o[slot * 5 + 4] = tb[i * 4 + 3];
                ++slot;
            }
        }
    }
}

extern "C" void kernel_launch(void* const* d_in, const int* in_sizes, int n_in,
                              void* d_out, int out_size, void* d_ws, size_t ws_size,
                              hipStream_t stream) {
    const float* im_data = (const float*)d_in[0];
    const float* im_info = (const float*)d_in[1];
    const float* conv_w  = (const float*)d_in[2];
    const float* conv_b  = (const float*)d_in[3];
    const float* cls_w   = (const float*)d_in[4];
    const float* cls_b   = (const float*)d_in[5];
    const float* bbox_w  = (const float*)d_in[6];
    const float* bbox_b  = (const float*)d_in[7];
    float* out = (float*)d_out;

    char* ws = (char*)d_ws;
    auto alloc = [&](size_t bytes) {
        char* p = ws;
        ws += (bytes + 255) & ~(size_t)255;
        return (void*)p;
    };
    unsigned short* inT  = (unsigned short*)alloc(8L * HWP * HWP * C_IN * 2);   // 35.7 MB
    unsigned short* w1   = (unsigned short*)alloc((long)C_MID * KTOT * 2);      //  4.7 MB
    unsigned short* xbf  = (unsigned short*)alloc(8L * C_MID * MPB * 2);        // 33.6 MB
    float*    scores     = (float*)alloc(8L * NPROP * 4);
    float*    boxes      = (float*)alloc(8L * NPROP * 16);
    int*      T          = (int*)alloc(8 * 4);
    int*      candCount  = (int*)alloc(8 * 4);
    int*      cand       = (int*)alloc(8L * NPROP * 4);
    float*    topScore   = (float*)alloc(8L * PRE_N * 4);
    float*    topBox     = (float*)alloc(8L * PRE_N * 16);
    unsigned* rowmask    = (unsigned*)alloc(8L * PRE_N * 63 * 4);

    long nq = (8L * HWP * HWP * C_IN) / 8;   // Quad count of inT
    k_zero<<<(int)((nq + 255) / 256), 256, 0, stream>>>((Quad*)inT, nq);
    k_tr<<<8 * HW, 256, 0, stream>>>(im_data, inT);
    long wTotal = (long)C_MID * KTOT;
    k_w1<<<(int)((wTotal + 255) / 256), 256, 0, stream>>>(conv_w, w1, wTotal);
    k_conv1<<<2048, 256, 0, stream>>>(inT, w1, conv_b, xbf);
    k_conv2<<<128, 256, 0, stream>>>(xbf, cls_w, cls_b, bbox_w, bbox_b, im_info, scores, boxes);
    k_hist<<<8, 1024, 0, stream>>>(scores, T, candCount);
    k_compact<<<8 * 144, 256, 0, stream>>>(scores, T, candCount, cand);
    k_rank<<<8 * 144, 256, 0, stream>>>(scores, boxes, candCount, cand, topScore, topBox);
    k_iou<<<64, 256, 0, stream>>>(topBox, rowmask);
    k_scan<<<8, 64, 0, stream>>>(rowmask, topScore, topBox, out);
}
